// HGFM_7954279432700
// MI455X (gfx1250) — matrix operand order repaired and checked by reading, not compiled, not run on hardware
//
#include <hip/hip_runtime.h>
#include <hip/hip_bf16.h>

#define NB   8
#define NN   512
#define N1   256
#define DDIM 64
#define TEMP_INV (1.0f/100.0f)

typedef _Float16 half8  __attribute__((ext_vector_type(8)));
typedef _Float16 v16h   __attribute__((ext_vector_type(16)));
typedef float    v8f    __attribute__((ext_vector_type(8)));

union V16U { v16h v; half8 h[2]; };

__device__ __forceinline__ float fast_tanh(float x) {
#if defined(__has_builtin)
#if __has_builtin(__builtin_amdgcn_tanhf)
    return __builtin_amdgcn_tanhf(x);
#else
    return tanhf(x);
#endif
#else
    return tanhf(x);
#endif
}

// ---------------------------------------------------------------------------
// broadcast stack node -> master0 (B,64)
__global__ __launch_bounds__(256) void k_bcast(const float* __restrict__ s,
                                               float* __restrict__ m0) {
    int gid = blockIdx.x * 256 + threadIdx.x;
    if (gid < NB * DDIM) m0[gid] = s[gid & 63];
}

// ---------------------------------------------------------------------------
// Xc[b,n,:] = (n<N1 ? x1@Wt1+bt1 : x2@Wt2+bt2); also f16 copy for WMMA A-matrix
__global__ __launch_bounds__(256) void k_proj(
    const float* __restrict__ s1, int bs1,
    const float* __restrict__ s2, int bs2,
    const float* __restrict__ Wt1, const float* __restrict__ bt1,
    const float* __restrict__ Wt2, const float* __restrict__ bt2,
    float* __restrict__ XcF, _Float16* __restrict__ XcH) {
    int gid = blockIdx.x * 256 + threadIdx.x;      // B*512*64
    int k = gid & 63;
    int n = (gid >> 6) & (NN - 1);
    int b = gid >> 15;
    const float *src, *W, *bias;
    if (n < N1) { src = s1 + (size_t)b * bs1 + n * DDIM;        W = Wt1; bias = bt1; }
    else        { src = s2 + (size_t)b * bs2 + (n - N1) * DDIM; W = Wt2; bias = bt2; }
    float acc = bias[k];
#pragma unroll 8
    for (int d = 0; d < DDIM; ++d) acc += src[d] * W[d * DDIM + k];
    XcF[gid] = acc;
    XcH[gid] = (_Float16)acc;
}

// ---------------------------------------------------------------------------
// The big fused kernel: one workgroup per (b,i).
//  S[j,k] = sum_d X[b,j,d] * (x_i[d]*Wap[d,k])   via v_wmma_f32_16x16x32_f16
//  logits[j] = sum_k w_sel[k] * tanh(S[j,k]+bap[k]);  softmax over j (temp 100)
//  agg[b,i,:] = sum_j att[j]*X[b,j,:];  optionally emit att row (layer-2 attmap)
__global__ __launch_bounds__(128) void k_pair(
    const float* __restrict__ XcF, const _Float16* __restrict__ XcH,
    const float* __restrict__ Wap, const float* __restrict__ bap,
    const float* __restrict__ w11, const float* __restrict__ w12,
    const float* __restrict__ w22,
    float* __restrict__ agg, float* __restrict__ attOut) {
    __shared__ __align__(32) _Float16 Wc[DDIM * DDIM]; // column-major: Wc[kcol*64+d]
    __shared__ float lg[NN];
    __shared__ float red[128];

    const int b = blockIdx.x >> 9;
    const int i = blockIdx.x & (NN - 1);
    const int t = threadIdx.x;

    // build B-operand diag(x_i) * Wap, f16, column-major
    const float* xi = XcF + ((size_t)b * NN + i) * DDIM;
    for (int e = t; e < DDIM * DDIM; e += 128) {
        int d = e & 63, kc = e >> 6;
        Wc[kc * DDIM + d] = (_Float16)(xi[d] * Wap[d * DDIM + kc]);
    }
    __syncthreads();

    const int lane  = t & 31;
    const int wave  = t >> 5;
    const int m     = lane & 15;   // row/col within tile
    const int g     = lane >> 4;   // lane half
    const int halfK = g * 8;

    // per-lane k = kt*16 + m constants
    const float* wtp = (i < N1) ? w11 : w12;
    const float* wbp = (i < N1) ? w12 : w22;
    float bp[4], wt[4], wb[4];
#pragma unroll
    for (int kt = 0; kt < 4; ++kt) {
        int k = kt * 16 + m;
        bp[kt] = bap[k]; wt[kt] = wtp[k]; wb[kt] = wbp[k];
    }

    const _Float16* Xb = XcH + (size_t)b * NN * DDIM;

    for (int jj = 0; jj < 8; ++jj) {
        const int jb = (wave * 8 + jj) * 16;
        const bool top = (jb < N1);
        // A operand: X rows jb..jb+15 (contiguous K=16g..16g+15 per lane, same as B)
        const _Float16* rp = Xb + (size_t)(jb + m) * DDIM;
        V16U aLo, aHi;
        aLo.h[0] = *(const half8*)(rp + 2 * halfK);
        aLo.h[1] = *(const half8*)(rp + 8 + 2 * halfK);
        aHi.h[0] = *(const half8*)(rp + 32 + 2 * halfK);
        aHi.h[1] = *(const half8*)(rp + 40 + 2 * halfK);

        float acc[8];
#pragma unroll
        for (int r = 0; r < 8; ++r) acc[r] = 0.f;

#pragma unroll
        for (int kt = 0; kt < 4; ++kt) {
            // B operand: column kcol = kt*16+m, contiguous K=16g..16g+15 per lane
            const _Float16* wp = Wc + (kt * 16 + m) * DDIM;
            v16h bLo = *(const v16h*)(wp + 16 * g);
            v16h bHi = *(const v16h*)(wp + 32 + 16 * g);
            v8f c = {};
            c = __builtin_amdgcn_wmma_f32_16x16x32_f16(false, aLo.v, false, bLo,
                                                       (short)0, c, false, false);
            c = __builtin_amdgcn_wmma_f32_16x16x32_f16(false, aHi.v, false, bHi,
                                                       (short)0, c, false, false);
            const float wk = top ? wt[kt] : wb[kt];
            const float bk = bp[kt];
#pragma unroll
            for (int r = 0; r < 8; ++r) acc[r] += wk * fast_tanh(c[r] + bk);
        }
        // reduce the 16 k-columns held across the 16 lanes of each half-wave
#pragma unroll
        for (int off = 1; off < 16; off <<= 1)
#pragma unroll
            for (int r = 0; r < 8; ++r) acc[r] += __shfl_xor(acc[r], off, 16);
        if (m == 0)
#pragma unroll
            for (int r = 0; r < 8; ++r) lg[jb + r + 8 * g] = acc[r];
    }
    __syncthreads();

    // softmax over j (512) with temperature 100
    float mx = -1e30f;
    for (int j = t; j < NN; j += 128) mx = fmaxf(mx, lg[j]);
    red[t] = mx; __syncthreads();
    for (int s = 64; s > 0; s >>= 1) {
        if (t < s) red[t] = fmaxf(red[t], red[t + s]);
        __syncthreads();
    }
    mx = red[0]; __syncthreads();
    float sm = 0.f;
    for (int j = t; j < NN; j += 128) {
        float e = __expf((lg[j] - mx) * TEMP_INV);
        lg[j] = e; sm += e;
    }
    red[t] = sm; __syncthreads();
    for (int s = 64; s > 0; s >>= 1) {
        if (t < s) red[t] += red[t + s];
        __syncthreads();
    }
    const float inv = 1.0f / red[0];
    __syncthreads();
    for (int j = t; j < NN; j += 128) lg[j] *= inv;
    __syncthreads();

    if (attOut) {
        float* ao = attOut + ((size_t)b * NN + i) * NN;
        for (int j = t; j < NN; j += 128) ao[j] = lg[j];
    }

    // agg[b,i,d] = sum_j att[j] * X[b,j,d]
    const int d = t & 63, half = t >> 6;
    const float* Xf = XcF + (size_t)b * NN * DDIM;
    float a = 0.f;
    for (int j = half * 256; j < half * 256 + 256; ++j) a += lg[j] * Xf[(size_t)j * DDIM + d];
    red[t] = a; __syncthreads();
    if (t < 64) agg[((size_t)b * NN + i) * DDIM + t] = red[t] + red[t + 64];
}

// ---------------------------------------------------------------------------
// master path: one block per batch, 512 threads (one per node)
__global__ __launch_bounds__(512) void k_master(
    const float* __restrict__ XcF, const float* __restrict__ mIn,
    const float* __restrict__ WapM, const float* __restrict__ bapM,
    const float* __restrict__ wM,
    const float* __restrict__ WwaM, const float* __restrict__ bwaM,
    const float* __restrict__ WwoM, const float* __restrict__ bwoM,
    float* __restrict__ mOut) {
    __shared__ float mL[DDIM];
    __shared__ float lg[NN];
    __shared__ float red[NN];
    __shared__ float am[DDIM];
    const int b = blockIdx.x, n = threadIdx.x;
    if (n < DDIM) mL[n] = mIn[b * DDIM + n];
    __syncthreads();

    const float* row = XcF + ((size_t)b * NN + n) * DDIM;
    float s[DDIM];
#pragma unroll
    for (int k = 0; k < DDIM; ++k) s[k] = bapM[k];
    for (int d = 0; d < DDIM; ++d) {
        const float u = row[d] * mL[d];
        const float* wr = WapM + d * DDIM;
#pragma unroll 16
        for (int k = 0; k < DDIM; ++k) s[k] += u * wr[k];
    }
    float l = 0.f;
#pragma unroll 16
    for (int k = 0; k < DDIM; ++k) l += wM[k] * fast_tanh(s[k]);

    // softmax over the 512 nodes
    red[n] = l; __syncthreads();
    for (int st = 256; st > 0; st >>= 1) {
        if (n < st) red[n] = fmaxf(red[n], red[n + st]);
        __syncthreads();
    }
    const float mx = red[0]; __syncthreads();
    const float e = __expf((l - mx) * TEMP_INV);
    red[n] = e; __syncthreads();
    for (int st = 256; st > 0; st >>= 1) {
        if (n < st) red[n] += red[n + st];
        __syncthreads();
    }
    lg[n] = e / red[0];
    __syncthreads();

    if (n < DDIM) {
        float a = 0.f;
        for (int j = 0; j < NN; ++j) a += lg[j] * XcF[((size_t)b * NN + j) * DDIM + n];
        am[n] = a;
    }
    __syncthreads();
    if (n < DDIM) {
        const int k = n;
        float o = bwaM[k] + bwoM[k];
        for (int d = 0; d < DDIM; ++d)
            o += am[d] * WwaM[d * DDIM + k] + mL[d] * WwoM[d * DDIM + k];
        mOut[b * DDIM + k] = o;
    }
}

// ---------------------------------------------------------------------------
// y = agg@Wwa + Xc@Wwo + bwa + bwo
__global__ __launch_bounds__(256) void k_y(
    const float* __restrict__ agg, const float* __restrict__ XcF,
    const float* __restrict__ Wwa, const float* __restrict__ bwa,
    const float* __restrict__ Wwo, const float* __restrict__ bwo,
    float* __restrict__ y) {
    int gid = blockIdx.x * 256 + threadIdx.x;
    int k = gid & 63;
    size_t rowb = (size_t)(gid >> 6) * DDIM;
    const float* ar = agg + rowb;
    const float* xr = XcF + rowb;
    float acc = bwa[k] + bwo[k];
#pragma unroll 8
    for (int d = 0; d < DDIM; ++d)
        acc += ar[d] * Wwa[d * DDIM + k] + xr[d] * Wwo[d * DDIM + k];
    y[gid] = acc;
}

// BatchNorm stats over (B,N) per channel -> scale/shift
__global__ __launch_bounds__(256) void k_bnstats(
    const float* __restrict__ y, const float* __restrict__ gamma,
    const float* __restrict__ beta, float* __restrict__ bn) {
    __shared__ float s1[256], s2[256];
    const int c = blockIdx.x, t = threadIdx.x;
    float a = 0.f, q = 0.f;
    for (int r = t; r < NB * NN; r += 256) {
        float v = y[(size_t)r * DDIM + c];
        a += v; q += v * v;
    }
    s1[t] = a; s2[t] = q; __syncthreads();
    for (int s = 128; s > 0; s >>= 1) {
        if (t < s) { s1[t] += s1[t + s]; s2[t] += s2[t + s]; }
        __syncthreads();
    }
    if (t == 0) {
        const float inv_n = 1.0f / (NB * NN);
        float mean = s1[0] * inv_n;
        float var  = s2[0] * inv_n - mean * mean;
        float sc   = gamma[c] * rsqrtf(var + 1e-5f);
        bn[c] = sc;
        bn[DDIM + c] = beta[c] - mean * sc;
    }
}

__global__ __launch_bounds__(256) void k_bnselu(
    const float* __restrict__ y, const float* __restrict__ bn,
    float* __restrict__ Xout) {
    int gid = blockIdx.x * 256 + threadIdx.x;
    int c = gid & 63;
    float v = y[gid] * bn[c] + bn[DDIM + c];
    const float SC = 1.0507009873554805f, AL = 1.6732632423543772f;
    Xout[gid] = v > 0.f ? SC * v : SC * AL * (__expf(v) - 1.f);
}

// out = (x1l1+x1l2) + (x2l1+x2l2)   (B,256,64)
__global__ __launch_bounds__(256) void k_final_nodes(
    const float* __restrict__ X1o, const float* __restrict__ X2o,
    float* __restrict__ out) {
    int gid = blockIdx.x * 256 + threadIdx.x;  // 8*256*64
    int b = gid >> 14;
    int r = gid & 16383;
    size_t base = (size_t)b * NN * DDIM;
    out[gid] = X1o[base + r] + X1o[base + 16384 + r] +
               X2o[base + r] + X2o[base + 16384 + r];
}

__global__ __launch_bounds__(256) void k_final_master(
    const float* __restrict__ m1, const float* __restrict__ m2,
    float* __restrict__ out) {
    int t = blockIdx.x * 256 + threadIdx.x;
    if (t < NB * DDIM) out[t] = m1[t] + m2[t];
}

// ---------------------------------------------------------------------------
struct LParams {
    const float *Wt1, *bt1, *Wt2, *bt2, *Wap, *bap, *WapM, *bapM,
                *w11, *w22, *w12, *wM, *Wwa, *bwa, *Wwo, *bwo,
                *WwaM, *bwaM, *WwoM, *bwoM, *gamma, *beta;
};

extern "C" void kernel_launch(void* const* d_in, const int* in_sizes, int n_in,
                              void* d_out, int out_size, void* d_ws, size_t ws_size,
                              hipStream_t stream) {
    (void)in_sizes; (void)n_in; (void)out_size; (void)ws_size;
    const float* x1 = (const float*)d_in[0];
    const float* x2 = (const float*)d_in[1];
    const float* stack = (const float*)d_in[2];

    LParams P1, P2;
    {
        const float** f1 = (const float**)&P1;
        const float** f2 = (const float**)&P2;
        for (int q = 0; q < 22; ++q) {
            f1[q] = (const float*)d_in[3 + q];
            f2[q] = (const float*)d_in[25 + q];
        }
    }

    // workspace bump allocation
    char* w = (char*)d_ws;
    auto alloc = [&](size_t bytes) -> void* {
        void* p = (void*)w;
        w += (bytes + 255) & ~(size_t)255;
        return p;
    };
    const size_t NE = (size_t)NB * NN * DDIM;  // 262144
    float*     XcF  = (float*)alloc(NE * 4);
    _Float16*  XcH  = (_Float16*)alloc(NE * 2);
    float*     aggB = (float*)alloc(NE * 4);
    float*     yB   = (float*)alloc(NE * 4);
    float*     X1o  = (float*)alloc(NE * 4);
    float*     X2o  = (float*)alloc(NE * 4);
    float*     m0   = (float*)alloc(NB * DDIM * 4);
    float*     m1   = (float*)alloc(NB * DDIM * 4);
    float*     m2   = (float*)alloc(NB * DDIM * 4);
    float*     bn   = (float*)alloc(2 * DDIM * 4);

    float* out     = (float*)d_out;
    float* mout    = out + (size_t)NB * N1 * DDIM;       // 131072
    float* attmap  = mout + (size_t)NB * DDIM;           // 131584

    const int GP = (int)(NE / 256);  // 1024 blocks for per-element grids

    // ---- Layer 1 ----
    k_bcast<<<2, 256, 0, stream>>>(stack, m0);
    k_proj<<<GP, 256, 0, stream>>>(x1, N1 * DDIM, x2, N1 * DDIM,
                                   P1.Wt1, P1.bt1, P1.Wt2, P1.bt2, XcF, XcH);
    k_pair<<<NB * NN, 128, 0, stream>>>(XcF, XcH, P1.Wap, P1.bap,
                                        P1.w11, P1.w12, P1.w22, aggB, nullptr);
    k_master<<<NB, 512, 0, stream>>>(XcF, m0, P1.WapM, P1.bapM, P1.wM,
                                     P1.WwaM, P1.bwaM, P1.WwoM, P1.bwoM, m1);
    k_y<<<GP, 256, 0, stream>>>(aggB, XcF, P1.Wwa, P1.bwa, P1.Wwo, P1.bwo, yB);
    k_bnstats<<<DDIM, 256, 0, stream>>>(yB, P1.gamma, P1.beta, bn);
    k_bnselu<<<GP, 256, 0, stream>>>(yB, bn, X1o);

    // ---- Layer 2 ----
    k_proj<<<GP, 256, 0, stream>>>(X1o, NN * DDIM, X1o + N1 * DDIM, NN * DDIM,
                                   P2.Wt1, P2.bt1, P2.Wt2, P2.bt2, XcF, XcH);
    k_pair<<<NB * NN, 128, 0, stream>>>(XcF, XcH, P2.Wap, P2.bap,
                                        P2.w11, P2.w12, P2.w22, aggB, attmap);
    k_master<<<NB, 512, 0, stream>>>(XcF, m1, P2.WapM, P2.bapM, P2.wM,
                                     P2.WwaM, P2.bwaM, P2.WwoM, P2.bwoM, m2);
    k_y<<<GP, 256, 0, stream>>>(aggB, XcF, P2.Wwa, P2.bwa, P2.Wwo, P2.bwo, yB);
    k_bnstats<<<DDIM, 256, 0, stream>>>(yB, P2.gamma, P2.beta, bn);
    k_bnselu<<<GP, 256, 0, stream>>>(yB, bn, X2o);

    // ---- Final combine ----
    k_final_nodes<<<(NB * N1 * DDIM) / 256, 256, 0, stream>>>(X1o, X2o, out);
    k_final_master<<<2, 256, 0, stream>>>(m1, m2, mout);
}
